// LRLA_4002909520011
// MI455X (gfx1250) — matrix operand order repaired and checked by reading, not compiled, not run on hardware
//
#include <hip/hip_runtime.h>
#include <math.h>

// ---------------------------------------------------------------------------
// Problem constants (from reference): B=16, H=W=56, C=192, NH=6
// ---------------------------------------------------------------------------
constexpr int Bn  = 16;
constexpr int Hh  = 56;
constexpr int Ww  = 56;
constexpr int Cc  = 192;
constexpr int NHh = 6;
constexpr int Nn  = Hh * Ww;        // 3136
constexpr int HD  = Cc / NHh;       // 32
constexpr int Mrows = Bn * Nn;      // 50176  (divisible by 128)

typedef __attribute__((ext_vector_type(16))) __bf16     v16bf;
typedef __attribute__((ext_vector_type(8)))  float      v8f;
typedef __attribute__((ext_vector_type(4)))  unsigned   u32x4;   // trivially constructible 16B chunk

union FragBF { u32x4 q[2]; v16bf v; };     // 16 bf16 = one WMMA A/B operand (8 VGPRs)

enum { ACT_NONE = 0, ACT_SILU = 1, ACT_GELU = 2, ACT_ELU1 = 3 };

__device__ __forceinline__ unsigned short f32_to_bf16(float f) {
  union { float f; unsigned u; } w; w.f = f;
  unsigned u = w.u;
  u += 0x7FFFu + ((u >> 16) & 1u);          // round to nearest even
  return (unsigned short)(u >> 16);
}

template <int ACT> __device__ __forceinline__ float apply_act(float x) {
  if (ACT == ACT_SILU) return x / (1.f + expf(-x));
  if (ACT == ACT_GELU) return 0.5f * x * (1.f + erff(x * 0.70710678118654752f));
  if (ACT == ACT_ELU1) return (x > 0.f) ? (x + 1.f) : expf(x);   // elu(x)+1
  return x;
}

// ---------------------------------------------------------------------------
// WMMA bf16 GEMM:  out[M,N] = act(A[M,K] @ W[K,N] + bias) (+ residual)
// A: [M,K] bf16 row-major. Bt: [N,K] bf16 (pre-transposed weights).
// Block: 256 threads = 8 waves (4x2), 128x64 tile; wave tile 32x32 = 2x2 WMMA.
// K-step 64 (8 WMMAs per wave per barrier pair), register double-buffered.
// ---------------------------------------------------------------------------
template <int ACT>
__global__ __launch_bounds__(256)
void gemm_bf16(const unsigned short* __restrict__ A,
               const unsigned short* __restrict__ Bt,
               const float* __restrict__ bias,
               const float* __restrict__ resid,
               float* __restrict__ outF,
               unsigned short* __restrict__ outH,
               int K, int N)
{
  __shared__ unsigned short sA[128 * 64];   // 16 KB, row stride 64 bf16 (128B)
  __shared__ unsigned short sB[64 * 72];    // 9 KB, [col][k] stride 72 (16B-aligned cols)

  const int tid  = threadIdx.x;
  const int lane = tid & 31;
  const int wave = tid >> 5;
  const int wm   = wave & 3;                // 0..3 -> M offset wm*32
  const int wn   = wave >> 2;               // 0..1 -> N offset wn*32
  const int blockM = blockIdx.x * 128;
  const int blockN = blockIdx.y * 64;

  // staging assignments (per thread): A: 64B of one row; B: 32B of one column
  const int arow  = tid >> 1;               // 0..127
  const int ahalf = (tid & 1) * 32;         // bf16 offset (32 elems = 64B)
  const unsigned short* aSrc = A  + (size_t)(blockM + arow) * K + ahalf;
  const int bcol  = tid >> 2;               // 0..63
  const int bseg  = (tid & 3) * 16;         // 16 bf16 = 32B
  const unsigned short* bSrc = Bt + (size_t)(blockN + bcol) * K + bseg;

  u32x4 ar[4], br[2];
  ar[0] = ((const u32x4*)aSrc)[0]; ar[1] = ((const u32x4*)aSrc)[1];
  ar[2] = ((const u32x4*)aSrc)[2]; ar[3] = ((const u32x4*)aSrc)[3];
  br[0] = ((const u32x4*)bSrc)[0]; br[1] = ((const u32x4*)bSrc)[1];

  v8f zero = 0.f;
  v8f acc[2][2];
  acc[0][0] = zero; acc[0][1] = zero; acc[1][0] = zero; acc[1][1] = zero;

  const int ra0 = wm * 32 + (lane & 15);
  const int ka  = (lane < 16) ? 0 : 8;      // A: lanes 0-15 K{0..7,16..23}, 16-31 K{8..15,24..31}
  const int cb0 = wn * 32 + (lane & 15);
  const int kb  = (lane < 16) ? 0 : 8;      // B: lanes 0-15 K{0..7,16..23}, 16-31 K{8..15,24..31}

  for (int k0 = 0;; k0 += 64) {
    __syncthreads();
    { u32x4* d = (u32x4*)(sA + arow * 64 + ahalf);
      d[0] = ar[0]; d[1] = ar[1]; d[2] = ar[2]; d[3] = ar[3]; }
    { u32x4* d = (u32x4*)(sB + bcol * 72 + bseg);
      d[0] = br[0]; d[1] = br[1]; }
    __syncthreads();

    const bool more = (k0 + 64) < K;
    if (more) {   // issue next tile's global loads early; they retire under the WMMAs
      const unsigned short* s = aSrc + (k0 + 64);
      ar[0] = ((const u32x4*)s)[0]; ar[1] = ((const u32x4*)s)[1];
      ar[2] = ((const u32x4*)s)[2]; ar[3] = ((const u32x4*)s)[3];
      const unsigned short* p = bSrc + (k0 + 64);
      br[0] = ((const u32x4*)p)[0]; br[1] = ((const u32x4*)p)[1];
    }

    #pragma unroll
    for (int kk = 0; kk < 64; kk += 32) {
      #pragma unroll
      for (int fm = 0; fm < 2; ++fm) {
        const int ra = ra0 + fm * 16;
        FragBF af;
        af.q[0] = *(const u32x4*)(sA + ra * 64 + kk + ka);
        af.q[1] = *(const u32x4*)(sA + ra * 64 + kk + ka + 16);
        #pragma unroll
        for (int fn = 0; fn < 2; ++fn) {
          const int cb = cb0 + fn * 16;
          FragBF bfr;
          bfr.q[0] = *(const u32x4*)(sB + cb * 72 + kk + kb);
          bfr.q[1] = *(const u32x4*)(sB + cb * 72 + kk + kb + 16);
          acc[fm][fn] = __builtin_amdgcn_wmma_f32_16x16x32_bf16(
              false, af.v, false, bfr.v, (short)0, acc[fm][fn], false, false);
        }
      }
    }
    if (!more) break;
  }

  // Epilogue: C/D layout — VGPR r, lane l: M = r + 8*(l>=16), N = l&15
  const int nlo = lane & 15;
  const int mhi = (lane >> 4) * 8;
  #pragma unroll
  for (int fm = 0; fm < 2; ++fm)
    #pragma unroll
    for (int fn = 0; fn < 2; ++fn)
      #pragma unroll
      for (int r = 0; r < 8; ++r) {
        const int row = blockM + wm * 32 + fm * 16 + mhi + r;
        const int col = blockN + wn * 32 + fn * 16 + nlo;
        const size_t o = (size_t)row * N + col;
        float vv = acc[fm][fn][r] + bias[col];
        vv = apply_act<ACT>(vv);
        if (resid) vv += resid[o];
        if (outF)  outF[o] = vv;
        if (outH)  outH[o] = f32_to_bf16(vv);
      }
}

// ---------------------------------------------------------------------------
// Depthwise 3x3 'SAME' conv:  v = act(conv(in)+bias) (+ residual)
// outF = v;  outH = bf16(v * mulsrc)  (mulsrc optional elementwise gate)
// ---------------------------------------------------------------------------
template <int ACT>
__global__ __launch_bounds__(256)
void dwconv3x3(const float* __restrict__ in, const float* __restrict__ w,
               const float* __restrict__ bias, const float* __restrict__ resid,
               const float* __restrict__ mulsrc,
               float* __restrict__ outF, unsigned short* __restrict__ outH, int Cch)
{
  const size_t t = (size_t)blockIdx.x * blockDim.x + threadIdx.x;
  const size_t total = (size_t)Bn * Hh * Ww * Cch;
  if (t >= total) return;
  const int c = (int)(t % Cch);
  size_t r = t / Cch;
  const int x = (int)(r % Ww); r /= Ww;
  const int y = (int)(r % Hh);
  const int b = (int)(r / Hh);
  float acc = bias[c];
  #pragma unroll
  for (int dy = 0; dy < 3; ++dy) {
    const int yy = y + dy - 1;
    if (yy < 0 || yy >= Hh) continue;
    #pragma unroll
    for (int dx = 0; dx < 3; ++dx) {
      const int xx = x + dx - 1;
      if (xx < 0 || xx >= Ww) continue;
      acc += in[(((size_t)b * Hh + yy) * Ww + xx) * Cch + c] * w[(dy * 3 + dx) * Cch + c];
    }
  }
  float v = apply_act<ACT>(acc);
  if (resid) v += resid[t];
  if (outF)  outF[t] = v;
  if (outH) {
    const float m = mulsrc ? v * mulsrc[t] : v;
    outH[t] = f32_to_bf16(m);
  }
}

// ---------------------------------------------------------------------------
// LayerNorm: one wave32 per row
// ---------------------------------------------------------------------------
template <int P>
__device__ __forceinline__ void ln_pass(float (&v)[P], const float* __restrict__ g,
                                        const float* __restrict__ be, int lane) {
  constexpr int L = P * 32;
  float s = 0.f;
  #pragma unroll
  for (int i = 0; i < P; ++i) s += v[i];
  #pragma unroll
  for (int m = 16; m >= 1; m >>= 1) s += __shfl_xor(s, m, 32);
  const float mean = s * (1.f / L);
  float q = 0.f;
  #pragma unroll
  for (int i = 0; i < P; ++i) { const float d = v[i] - mean; q += d * d; }
  #pragma unroll
  for (int m = 16; m >= 1; m >>= 1) q += __shfl_xor(q, m, 32);
  const float inv = rsqrtf(q * (1.f / L) + 1e-5f);
  #pragma unroll
  for (int i = 0; i < P; ++i)
    v[i] = (v[i] - mean) * inv * g[lane + i * 32] + be[lane + i * 32];
}

template <int L, int ACT>
__global__ __launch_bounds__(256)
void ln_rows(const float* __restrict__ in, const float* __restrict__ add,
             const float* __restrict__ g, const float* __restrict__ be,
             float* __restrict__ outF, unsigned short* __restrict__ outH, int rows)
{
  const int lane = threadIdx.x & 31;
  const int row  = blockIdx.x * 8 + (threadIdx.x >> 5);
  if (row >= rows) return;
  constexpr int P = L / 32;
  float v[P];
  const size_t base = (size_t)row * L;
  #pragma unroll
  for (int i = 0; i < P; ++i) {
    float t = in[base + lane + i * 32];
    if (add) t += add[base + lane + i * 32];
    v[i] = t;
  }
  ln_pass<P>(v, g, be, lane);
  #pragma unroll
  for (int i = 0; i < P; ++i) {
    float o = apply_act<ACT>(v[i]);
    if (outF) outF[base + lane + i * 32] = o;
    if (outH) outH[base + lane + i * 32] = f32_to_bf16(o);
  }
}

// Fused MLP tail: n1=LN(dw+f1); n2=LN(n1+f1); n3=LN(n2+f1); out=bf16(gelu(n3))
__global__ __launch_bounds__(256)
void ln_chain768(const float* __restrict__ dw, const float* __restrict__ f1,
                 const float* __restrict__ g1, const float* __restrict__ b1,
                 const float* __restrict__ g2, const float* __restrict__ b2,
                 const float* __restrict__ g3, const float* __restrict__ b3,
                 unsigned short* __restrict__ outH, int rows)
{
  constexpr int P = 768 / 32;
  const int lane = threadIdx.x & 31;
  const int row  = blockIdx.x * 8 + (threadIdx.x >> 5);
  if (row >= rows) return;
  const size_t base = (size_t)row * 768;
  float f[P], v[P];
  #pragma unroll
  for (int i = 0; i < P; ++i) {
    f[i] = f1[base + lane + i * 32];
    v[i] = dw[base + lane + i * 32] + f[i];
  }
  ln_pass<P>(v, g1, b1, lane);
  #pragma unroll
  for (int i = 0; i < P; ++i) v[i] += f[i];
  ln_pass<P>(v, g2, b2, lane);
  #pragma unroll
  for (int i = 0; i < P; ++i) v[i] += f[i];
  ln_pass<P>(v, g3, b3, lane);
  #pragma unroll
  for (int i = 0; i < P; ++i)
    outH[base + lane + i * 32] = f32_to_bf16(apply_act<ACT_GELU>(v[i]));
}

// ---------------------------------------------------------------------------
// Elementwise / attention helpers
// ---------------------------------------------------------------------------
// Convert fp32 weight [K,N] -> bf16 transposed [N,K] (once per call, tiny)
__global__ __launch_bounds__(256)
void cvt_t_bf16(const float* __restrict__ a, unsigned short* __restrict__ o, int K, int N) {
  const int t = blockIdx.x * blockDim.x + threadIdx.x;
  if (t >= K * N) return;
  const int k = t / N, n = t % N;
  o[(size_t)n * K + k] = f32_to_bf16(a[t]);
}

// RoPE over q and k halves of qk buffer [B,N,2C] (pairwise rotate)
__global__ __launch_bounds__(256)
void rope_qk(const float* __restrict__ qk, float* __restrict__ out) {
  const size_t t = (size_t)blockIdx.x * blockDim.x + threadIdx.x;
  if (t >= (size_t)Bn * Nn * Cc) return;
  const int s = (int)(t % Cc);
  size_t r = t / Cc;
  const int n = (int)(r % Nn);
  const int b = (int)(r / Nn);
  const int which = s / (Cc / 2);                 // 0 = q, 1 = k
  const int p     = s % (Cc / 2);                 // pair index 0..95
  const int pos = (p < Cc / 4) ? (n / Ww) : (n % Ww);
  const int pi  = (p < Cc / 4) ? p : (p - Cc / 4);
  const float ang = pos * expf(-(float)pi * (9.210340371976184f / (Cc / 4))); // 1/10000^(pi/48)
  const float cs = cosf(ang), sn = sinf(ang);
  const size_t base = ((size_t)(b * Nn + n)) * (2 * Cc) + (size_t)which * Cc + 2 * p;
  const float xr = qk[base], xi = qk[base + 1];
  out[base]     = cs * xr - sn * xi;
  out[base + 1] = cs * xi + sn * xr;
}

// kmean[b,h,d] = mean over N of k (non-rotated, second half of qk buffer)
__global__ __launch_bounds__(256)
void kmean_kernel(const float* __restrict__ qk, float* __restrict__ km) {
  const int t = blockIdx.x * blockDim.x + threadIdx.x;
  if (t >= Bn * NHh * HD) return;
  const int d = t % HD;
  const int h = (t / HD) % NHh;
  const int b = t / (HD * NHh);
  float s = 0.f;
  for (int n = 0; n < Nn; ++n)
    s += qk[((size_t)(b * Nn + n)) * (2 * Cc) + Cc + h * HD + d];
  km[t] = s * (1.f / Nn);
}

// kv[b,h,d,e] = (1/N) * sum_n k_r[b,n,h,d] * v[b,n,h,e]   (LDS-tiled over n)
__global__ __launch_bounds__(1024)
void kv_kernel(const float* __restrict__ kr, const float* __restrict__ v,
               float* __restrict__ kv) {
  __shared__ float lk[32][33];
  __shared__ float lv[32][33];
  const int h = blockIdx.x % NHh;
  const int b = blockIdx.x / NHh;
  const int d = threadIdx.x >> 5;
  const int e = threadIdx.x & 31;
  float acc = 0.f;
  for (int n0 = 0; n0 < Nn; n0 += 32) {
    __syncthreads();
    const int nn = threadIdx.x >> 5, j = threadIdx.x & 31;
    lk[nn][j] = kr[((size_t)(b * Nn + n0 + nn)) * (2 * Cc) + Cc + h * HD + j];
    lv[nn][j] = v [((size_t)(b * Nn + n0 + nn)) * Cc + h * HD + j];
    __syncthreads();
    #pragma unroll
    for (int k = 0; k < 32; ++k) acc += lk[k][d] * lv[k][e];
  }
  kv[((size_t)blockIdx.x * HD + d) * HD + e] = acc * (1.f / Nn);
}

// attn[b,n,c] = z * sum_d q_r[b,n,h,d]*kv[b,h,d,e],  z = 1/(q·kmean + 1e-6)
__global__ __launch_bounds__(256)
void attn_kernel(const float* __restrict__ qk, const float* __restrict__ qr,
                 const float* __restrict__ kv, const float* __restrict__ km,
                 float* __restrict__ out) {
  const size_t t = (size_t)blockIdx.x * blockDim.x + threadIdx.x;
  if (t >= (size_t)Bn * Nn * Cc) return;
  const int c = (int)(t % Cc);
  size_t r = t / Cc;
  const int n = (int)(r % Nn);
  const int b = (int)(r / Nn);
  const int h = c / HD, e = c % HD;
  const size_t qbase = ((size_t)(b * Nn + n)) * (2 * Cc) + h * HD;
  const float* kmp = km + (size_t)(b * NHh + h) * HD;
  const float* kvp = kv + ((size_t)(b * NHh + h)) * HD * HD + e;
  float z = 0.f, a = 0.f;
  #pragma unroll
  for (int d = 0; d < HD; ++d) z += qk[qbase + d] * kmp[d];
  z = 1.f / (z + 1e-6f);
  #pragma unroll
  for (int d = 0; d < HD; ++d) a += qr[qbase + d] * kvp[(size_t)d * HD];
  out[t] = a * z;
}

// ---------------------------------------------------------------------------
// Host orchestration
// ---------------------------------------------------------------------------
extern "C" void kernel_launch(void* const* d_in, const int* in_sizes, int n_in,
                              void* d_out, int out_size, void* d_ws, size_t ws_size,
                              hipStream_t stream) {
  (void)in_sizes; (void)n_in; (void)out_size; (void)ws_size;

  const float* x_in    = (const float*)d_in[0];
  const float* cpe1_w  = (const float*)d_in[1];
  const float* cpe1_b  = (const float*)d_in[2];
  const float* norm1_g = (const float*)d_in[3];
  const float* norm1_b = (const float*)d_in[4];
  const float* in_w    = (const float*)d_in[5];
  const float* in_b    = (const float*)d_in[6];
  const float* actp_w  = (const float*)d_in[7];
  const float* actp_b  = (const float*)d_in[8];
  const float* dwc_w   = (const float*)d_in[9];
  const float* dwc_b   = (const float*)d_in[10];
  const float* qk_w    = (const float*)d_in[11];
  const float* qk_b    = (const float*)d_in[12];
  const float* lepe_w  = (const float*)d_in[13];
  const float* lepe_b  = (const float*)d_in[14];
  const float* outp_w  = (const float*)d_in[15];
  const float* outp_b  = (const float*)d_in[16];
  const float* cpe2_w  = (const float*)d_in[17];
  const float* cpe2_b  = (const float*)d_in[18];
  const float* norm2_g = (const float*)d_in[19];
  const float* norm2_b = (const float*)d_in[20];
  const float* fc1_w   = (const float*)d_in[21];
  const float* fc1_b   = (const float*)d_in[22];
  const float* mdw_w   = (const float*)d_in[23];
  const float* mdw_b   = (const float*)d_in[24];
  const float* fc2_w   = (const float*)d_in[25];
  const float* fc2_b   = (const float*)d_in[26];
  const float* mn1_g   = (const float*)d_in[27];
  const float* mn1_b   = (const float*)d_in[28];
  const float* mn2_g   = (const float*)d_in[29];
  const float* mn2_b   = (const float*)d_in[30];
  const float* mn3_g   = (const float*)d_in[31];
  const float* mn3_b   = (const float*)d_in[32];

  float* out = (float*)d_out;

  // Workspace layout (units: floats). S = B*N*C.
  const size_t S = (size_t)Bn * Nn * Cc;
  float* ws    = (float*)d_ws;
  float* bx    = ws;                 // [0,   S)  x residual (also head of 4S dw region)
  float* bact  = ws + S;             // [S,  2S)  act_res
  float* bh    = ws + 2 * S;         // [2S, 3S)  h
  float* bt    = ws + 3 * S;         // [3S, 4S)  in-proj tmp, later attn
  float* bqk   = ws + 4 * S;         // [4S, 6S)  elu'd q|k, later head of f1 (4S)
  float* brope = ws + 6 * S;         // [6S, 8S)  q_r|k_r
  float* bx2   = ws + 8 * S;         // [8S, 9S)  x after cpe2
  unsigned short* bfA = (unsigned short*)(ws + 9 * S);  // 4S bf16 activation staging
  float* bkv   = ws + 11 * S;                            // B*NH*HD*HD
  float* bkm   = bkv + (size_t)Bn * NHh * HD * HD;       // B*NH*HD
  unsigned short* wbf = (unsigned short*)(bkm + (size_t)Bn * NHh * HD);
  unsigned short* w_actp = wbf;                  // all bf16 transposed [N,K]
  unsigned short* w_in   = w_actp + Cc * Cc;
  unsigned short* w_qk   = w_in   + Cc * Cc;
  unsigned short* w_outp = w_qk   + Cc * 2 * Cc;
  unsigned short* w_fc1  = w_outp + Cc * Cc;
  unsigned short* w_fc2  = w_fc1  + Cc * 4 * Cc;
  float* f1  = bqk;   // [4S,8S) reuses qk+rope after attention
  float* bdw = bx;    // [0, 4S) reuses x/act/h/attn after MLP input is formed

  const int ln_blocks = Mrows / 8;                  // 6272
  const int ew192     = (int)(S / 256);             // 37632 blocks for B*N*C grids
  const dim3 g192(Mrows / 128, 192 / 64);
  const dim3 g384(Mrows / 128, 384 / 64);
  const dim3 g768(Mrows / 128, 768 / 64);

  // 0. weights -> bf16 transposed [N,K]
  cvt_t_bf16<<<(Cc * Cc + 255) / 256, 256, 0, stream>>>(actp_w, w_actp, Cc, Cc);
  cvt_t_bf16<<<(Cc * Cc + 255) / 256, 256, 0, stream>>>(in_w,   w_in,   Cc, Cc);
  cvt_t_bf16<<<(Cc * 2 * Cc + 255) / 256, 256, 0, stream>>>(qk_w, w_qk, Cc, 2 * Cc);
  cvt_t_bf16<<<(Cc * Cc + 255) / 256, 256, 0, stream>>>(outp_w, w_outp, Cc, Cc);
  cvt_t_bf16<<<(Cc * 4 * Cc + 255) / 256, 256, 0, stream>>>(fc1_w, w_fc1, Cc, 4 * Cc);
  cvt_t_bf16<<<(4 * Cc * Cc + 255) / 256, 256, 0, stream>>>(fc2_w, w_fc2, 4 * Cc, Cc);

  // 1. x = x + dwconv(x, cpe1)
  dwconv3x3<ACT_NONE><<<ew192, 256, 0, stream>>>(x_in, cpe1_w, cpe1_b, x_in, nullptr, bx, nullptr, Cc);
  // 2. xn = LN(x) -> bf16
  ln_rows<192, ACT_NONE><<<ln_blocks, 256, 0, stream>>>(bx, nullptr, norm1_g, norm1_b, nullptr, bfA, Mrows);
  // 3. act_res = silu(xn @ actp_w + b)
  gemm_bf16<ACT_SILU><<<g192, 256, 0, stream>>>(bfA, w_actp, actp_b, nullptr, bact, nullptr, 192, 192);
  // 4. t = xn @ in_w + b
  gemm_bf16<ACT_NONE><<<g192, 256, 0, stream>>>(bfA, w_in, in_b, nullptr, bt, nullptr, 192, 192);
  // 5. h = silu(dwconv(t, dwc) + b)   (f32 + bf16 copy)
  dwconv3x3<ACT_SILU><<<ew192, 256, 0, stream>>>(bt, dwc_w, dwc_b, nullptr, nullptr, bh, bfA, Cc);
  // 6. q|k = elu(h @ qk_w + b) + 1    (elu fused into GEMM epilogue)
  gemm_bf16<ACT_ELU1><<<g384, 256, 0, stream>>>(bfA, w_qk, qk_b, nullptr, bqk, nullptr, 192, 384);
  // 7. RoPE -> q_r|k_r
  rope_qk<<<ew192, 256, 0, stream>>>(bqk, brope);
  // 8. kmean
  kmean_kernel<<<(Bn * NHh * HD + 255) / 256, 256, 0, stream>>>(bqk, bkm);
  // 9. kv = k_r^T v / N
  kv_kernel<<<Bn * NHh, 1024, 0, stream>>>(brope, bh, bkv);
  // 10. attn = (q_r @ kv) * z  -> bt
  attn_kernel<<<ew192, 256, 0, stream>>>(bqk, brope, bkv, bkm, bt);
  // 11. m = (attn + dwconv(h, lepe) + b) * act_res -> bf16   (lepe + gate fused)
  dwconv3x3<ACT_NONE><<<ew192, 256, 0, stream>>>(bh, lepe_w, lepe_b, bt, bact, nullptr, bfA, Cc);
  // 12. x = shortcut + m @ outp_w + b   (residual fused, in-place on bx)
  gemm_bf16<ACT_NONE><<<g192, 256, 0, stream>>>(bfA, w_outp, outp_b, bx, bx, nullptr, 192, 192);
  // 13. x = x + dwconv(x, cpe2) -> bx2
  dwconv3x3<ACT_NONE><<<ew192, 256, 0, stream>>>(bx, cpe2_w, cpe2_b, bx, nullptr, bx2, nullptr, Cc);
  // 14. xm = LN(x) -> bf16
  ln_rows<192, ACT_NONE><<<ln_blocks, 256, 0, stream>>>(bx2, nullptr, norm2_g, norm2_b, nullptr, bfA, Mrows);
  // 15. f1 = xm @ fc1_w + b   (4S f32)
  gemm_bf16<ACT_NONE><<<g768, 256, 0, stream>>>(bfA, w_fc1, fc1_b, nullptr, f1, nullptr, 192, 768);
  // 16. dw = dwconv(f1, mdw) + b
  dwconv3x3<ACT_NONE><<<(int)(4 * S / 256), 256, 0, stream>>>(f1, mdw_w, mdw_b, nullptr, nullptr, bdw, nullptr, 4 * Cc);
  // 17. fused LN chain + gelu -> bf16
  ln_chain768<<<ln_blocks, 256, 0, stream>>>(bdw, f1, mn1_g, mn1_b, mn2_g, mn2_b, mn3_g, mn3_b, bfA, Mrows);
  // 18. out = x + gelu(n3) @ fc2_w + b
  gemm_bf16<ACT_NONE><<<g192, 256, 0, stream>>>(bfA, w_fc2, fc2_b, bx2, out, nullptr, 768, 192);
}
